// CustomMamba_76459007803471
// MI455X (gfx1250) — matrix operand order repaired and checked by reading, not compiled, not run on hardware
//
#include <hip/hip_runtime.h>
#include <hip/hip_bf16.h>

typedef _Float16 h16;
typedef __attribute__((ext_vector_type(16))) _Float16 v16h;
typedef __attribute__((ext_vector_type(8)))  float    v8f;

static constexpr int L_ = 2, B_ = 256, T_ = 64;
static constexpr int IN_ = 512, OUT_ = 512, DM_ = 512, DS_ = 16, DC_ = 4;
static constexpr int DI_ = 1024, DTR_ = 32;
static constexpr float EPS_ = 1e-5f;

#define WMMA_F16(a, b, c) \
    __builtin_amdgcn_wmma_f32_16x16x32_f16(false, (a), false, (b), (short)0, (c), false, false)

__device__ __forceinline__ float silu_f(float x) { return x / (1.f + __expf(-x)); }
__device__ __forceinline__ float softplus_f(float x) {
    return fmaxf(x, 0.f) + log1pf(__expf(-fabsf(x)));
}

// ---- WMMA fragment loaders (layouts per CDNA5 ISA 7.12.2, wave32) ----

// A fragment 16x32, rows row0..row0+15, K = k0..k0+31, from row-major (lda); T = float or h16
template <typename T>
__device__ __forceinline__ v16h load_a_any(const T* __restrict__ A, int lda, int row0, int k0) {
    int lane = threadIdx.x & 31;
    const T* p = A + (size_t)(row0 + (lane & 15)) * lda + k0 + ((lane < 16) ? 0 : 8);
    v16h a;
#pragma unroll
    for (int i = 0; i < 8; i++) a[i] = (h16)p[i];
#pragma unroll
    for (int i = 0; i < 8; i++) a[8 + i] = (h16)p[16 + i];
    return a;
}

// A fragment with fused residual-add + rmsnorm scaling. A/Bsrc are base+row0g*512.
template <bool HAS_B>
__device__ __forceinline__ v16h load_a_norm(const float* __restrict__ A,
                                            const float* __restrict__ Bsrc,
                                            const float* __restrict__ rstd,
                                            const float* __restrict__ nw,
                                            int mloc, int k0) {
    int lane = threadIdx.x & 31;
    int m = mloc + (lane & 15);
    int kb = (lane < 16) ? 0 : 8;
    const float* pa = A + (size_t)m * 512 + k0 + kb;
    const float* pb = Bsrc + (size_t)m * 512 + k0 + kb;
    const float* w = nw + k0 + kb;
    float s = rstd[m];
    v16h a;
#pragma unroll
    for (int i = 0; i < 8; i++) {
        float v = pa[i] + (HAS_B ? pb[i] : 0.f);
        a[i] = (h16)(v * s * w[i]);
    }
#pragma unroll
    for (int i = 0; i < 8; i++) {
        float v = pa[16 + i] + (HAS_B ? pb[16 + i] : 0.f);
        a[8 + i] = (h16)(v * s * w[16 + i]);
    }
    return a;
}

// B fragment 32x16 where B[k][n] = W[n][k], W row-major (N,K) f16, cols n0..n0+15
__device__ __forceinline__ v16h load_b_h16(const h16* __restrict__ W, int ldw, int n0, int k0) {
    int lane = threadIdx.x & 31;
    const h16* p = W + (size_t)(n0 + (lane & 15)) * ldw + k0 + ((lane < 16) ? 0 : 8);
    v16h b;
#pragma unroll
    for (int i = 0; i < 8; i++) b[i] = p[i];
#pragma unroll
    for (int i = 0; i < 8; i++) b[8 + i] = p[16 + i];
    return b;
}

// C fragment store (16x16), compile-time optional bias over columns
template <bool HAS_BIAS>
__device__ __forceinline__ void store_c(float* __restrict__ C, int ldc, int row0, int col0,
                                        v8f c, const float* __restrict__ bias) {
    int lane = threadIdx.x & 31;
    int n = col0 + (lane & 15);
    int mo = (lane < 16) ? 0 : 8;
    float bv = HAS_BIAS ? bias[n] : 0.f;
#pragma unroll
    for (int r = 0; r < 8; r++) C[(size_t)(row0 + r + mo) * ldc + n] = c[r] + bv;
}

// ---- small utility kernels ----

__global__ __launch_bounds__(256) void cvt_f16_kernel(const float* __restrict__ s,
                                                      h16* __restrict__ d, int n) {
    int i = blockIdx.x * 256 + threadIdx.x;
    if (i < n) d[i] = (h16)s[i];
}

__global__ __launch_bounds__(256) void aneg_kernel(const float* __restrict__ alog,
                                                   float* __restrict__ an, int n) {
    int i = blockIdx.x * 256 + threadIdx.x;
    if (i < n) an[i] = -__expf(alog[i]);
}

// ---- plain GEMM: C(M,N) = A(M,K) @ W(N,K)^T (+bias).
// Block 128 thr = 4 waves (2x2); each wave 32x64 (2x4 tiles) -> block tile 64x128.
template <typename TA, bool HAS_BIAS>
__global__ __launch_bounds__(128) void gemm_bias_kernel(const TA* __restrict__ A,
                                                        const h16* __restrict__ W,
                                                        const float* __restrict__ bias,
                                                        float* __restrict__ C, int K, int N) {
    int wid = threadIdx.x >> 5;
    int row0 = blockIdx.x * 64 + (wid >> 1) * 32;
    int col0 = blockIdx.y * 128 + (wid & 1) * 64;
    v8f acc[2][4];
#pragma unroll
    for (int i = 0; i < 2; i++)
#pragma unroll
        for (int j = 0; j < 4; j++) acc[i][j] = {};
    for (int k0 = 0; k0 < K; k0 += 32) {
        if (k0 + 32 < K)  // pull next K-slice of the weight stream toward the WGP
            __builtin_prefetch(W + (size_t)(col0 + (threadIdx.x & 31)) * K + k0 + 32, 0, 3);
        v16h a0 = load_a_any(A, K, row0, k0);
        v16h a1 = load_a_any(A, K, row0 + 16, k0);
        v16h bf[4];
#pragma unroll
        for (int j = 0; j < 4; j++) bf[j] = load_b_h16(W, K, col0 + j * 16, k0);
#pragma unroll
        for (int j = 0; j < 4; j++) {
            acc[0][j] = WMMA_F16(a0, bf[j], acc[0][j]);
            acc[1][j] = WMMA_F16(a1, bf[j], acc[1][j]);
        }
    }
#pragma unroll
    for (int i = 0; i < 2; i++)
#pragma unroll
        for (int j = 0; j < 4; j++)
            store_c<HAS_BIAS>(C, N, row0 + i * 16, col0 + j * 16, acc[i][j], bias);
}

// ---- fused (residual add) + RMSNorm + GEMM. K fixed at 512. Block tile 64x128.
// r = srcA (+ srcB); optionally write r to residOut; C = rmsnorm(r, normw) @ W^T (+bias)
template <bool HAS_B, bool HAS_ROUT, bool HAS_BIAS>
__global__ __launch_bounds__(128) void gemm_norm_kernel(const float* __restrict__ srcA,
                                                        const float* __restrict__ srcB,
                                                        float* __restrict__ residOut,
                                                        const float* __restrict__ normw,
                                                        const h16* __restrict__ W,
                                                        const float* __restrict__ bias,
                                                        float* __restrict__ C, int N) {
    __shared__ float nw[512];
    __shared__ float rstd[64];
    __shared__ float red[128];
    int tid = threadIdx.x;
    int row0g = blockIdx.x * 64;
    const float* A = srcA + (size_t)row0g * 512;
    const float* Bs = HAS_B ? srcB + (size_t)row0g * 512 : nullptr;
    {
        int r = tid >> 1, hh = tid & 1;
        const float* pa = A + (size_t)r * 512 + hh * 256;
        const float* pb = HAS_B ? Bs + (size_t)r * 512 + hh * 256 : nullptr;
        float* po = HAS_ROUT ? residOut + (size_t)(row0g + r) * 512 + hh * 256 : nullptr;
        float acc = 0.f;
        for (int c0 = 0; c0 < 256; c0++) {
            float v = pa[c0] + (HAS_B ? pb[c0] : 0.f);
            acc += v * v;
            if (HAS_ROUT) po[c0] = v;
        }
        red[tid] = acc;
    }
    for (int i = tid; i < 512; i += 128) nw[i] = normw[i];
    __syncthreads();
    if (tid < 64) rstd[tid] = rsqrtf((red[2 * tid] + red[2 * tid + 1]) * (1.f / 512.f) + EPS_);
    __syncthreads();

    int wid = tid >> 5;
    int m0 = (wid >> 1) * 32;
    int col0 = blockIdx.y * 128 + (wid & 1) * 64;
    v8f acc[2][4];
#pragma unroll
    for (int i = 0; i < 2; i++)
#pragma unroll
        for (int j = 0; j < 4; j++) acc[i][j] = {};
    for (int k0 = 0; k0 < 512; k0 += 32) {
        if (k0 + 32 < 512)
            __builtin_prefetch(W + (size_t)(col0 + (threadIdx.x & 31)) * 512 + k0 + 32, 0, 3);
        v16h a0 = load_a_norm<HAS_B>(A, Bs, rstd, nw, m0, k0);
        v16h a1 = load_a_norm<HAS_B>(A, Bs, rstd, nw, m0 + 16, k0);
        v16h bf[4];
#pragma unroll
        for (int j = 0; j < 4; j++) bf[j] = load_b_h16(W, 512, col0 + j * 16, k0);
#pragma unroll
        for (int j = 0; j < 4; j++) {
            acc[0][j] = WMMA_F16(a0, bf[j], acc[0][j]);
            acc[1][j] = WMMA_F16(a1, bf[j], acc[1][j]);
        }
    }
#pragma unroll
    for (int i = 0; i < 2; i++)
#pragma unroll
        for (int j = 0; j < 4; j++)
            store_c<HAS_BIAS>(C, N, row0g + m0 + i * 16, col0 + j * 16, acc[i][j], bias);
}

// ---- fused conv -> silu -> x_proj(WMMA) -> dt(WMMA)+softplus -> SSM recurrence -> gate
// One block handles 16 batch rows x full DI. Dynamic LDS: xi[16*1024], dt[16*1024], xdb[16*64]
__global__ __launch_bounds__(256) void ssm_kernel(const float* __restrict__ xz,
                                                  float* __restrict__ st_l,   // (B,DI,20)
                                                  const float* __restrict__ convw,
                                                  const float* __restrict__ convb,
                                                  const h16* __restrict__ xproj,   // (64,1024)
                                                  const h16* __restrict__ dtproj,  // (1024,32)
                                                  const float* __restrict__ dtb,
                                                  const float* __restrict__ An,    // (DI,16) = -exp(A_log)
                                                  const float* __restrict__ Dp,
                                                  h16* __restrict__ yg) {
    extern __shared__ float sm[];
    float* xi = sm;                  // 16*1024
    float* dt = xi + 16 * 1024;      // 16*1024
    float* xdb = dt + 16 * 1024;     // 16*64
    int tid = threadIdx.x;
    int b0 = blockIdx.x * 16;

    // phase 1: depthwise conv state shift + silu
    for (int idx = tid; idx < 16 * 1024; idx += 256) {
        int r = idx >> 10, d = idx & 1023, b = b0 + r;
        float xin = xz[(size_t)b * 2048 + d];
        float* s = st_l + ((size_t)b * 1024 + d) * 20;
        float c0 = s[1], c1 = s[2], c2 = s[3];
        s[0] = c0; s[1] = c1; s[2] = c2; s[3] = xin;
        const float* w = convw + (size_t)d * 4;
        float v = c0 * w[0] + c1 * w[1] + c2 * w[2] + xin * w[3] + convb[d];
        xi[idx] = silu_f(v);
    }
    __syncthreads();

    int wid = tid >> 5;
    // phase 2: xdb(16x64) = xi(16x1024) @ x_proj_w^T  (waves 0..3, one 16-col tile each)
    if (wid < 4) {
        int n0 = wid * 16;
        v8f c = {};
        for (int k0 = 0; k0 < 1024; k0 += 32) {
            v16h a = load_a_any(xi, 1024, 0, k0);
            v16h bf = load_b_h16(xproj, 1024, n0, k0);
            c = WMMA_F16(a, bf, c);
        }
        store_c<false>(xdb, 64, 0, n0, c, nullptr);
    }
    __syncthreads();

    // phase 3: dt(16x1024) = softplus(xdb[:, :32] @ dt_proj_w^T + dt_proj_b) (8 waves x 8 tiles)
    {
        v16h a = load_a_any(xdb, 64, 0, 0);  // K=32 == one WMMA step
        int lane = tid & 31;
        int nn = lane & 15, mo = (lane < 16) ? 0 : 8;
        for (int tix = 0; tix < 8; tix++) {
            int n0 = (wid * 8 + tix) * 16;
            v16h bf = load_b_h16(dtproj, 32, n0, 0);
            v8f c = {};
            c = WMMA_F16(a, bf, c);
            int n = n0 + nn;
            float bb = dtb[n];
#pragma unroll
            for (int r = 0; r < 8; r++) dt[(r + mo) * 1024 + n] = softplus_f(c[r] + bb);
        }
    }
    __syncthreads();

    // phase 4: SSM recurrence + output gate; yg emitted as f16 (WMMA A-operand downstream)
    for (int idx = tid; idx < 16 * 1024; idx += 256) {
        int r = idx >> 10, d = idx & 1023, b = b0 + r;
        float dtv = dt[idx];
        float xiv = xi[idx];
        float dx = dtv * xiv;
        const float* an = An + (size_t)d * 16;
        float* ss = st_l + ((size_t)b * 1024 + d) * 20 + 4;
        const float* bc = xdb + r * 64;
        float y = 0.f;
#pragma unroll
        for (int n = 0; n < 16; n++) {
            float s = ss[n] * __expf(dtv * an[n]) + dx * bc[32 + n];
            ss[n] = s;
            y += s * bc[48 + n];
        }
        y += Dp[d] * xiv;
        float z = xz[(size_t)b * 2048 + 1024 + d];
        yg[(size_t)b * 1024 + d] = (h16)(y * silu_f(z));
    }
}

extern "C" void kernel_launch(void* const* d_in, const int* in_sizes, int n_in,
                              void* d_out, int out_size, void* d_ws, size_t ws_size,
                              hipStream_t stream) {
    const float* x         = (const float*)d_in[0];
    const float* rnn       = (const float*)d_in[1];
    const float* inp_w     = (const float*)d_in[2];
    const float* inp_b     = (const float*)d_in[3];
    const float* outp_w    = (const float*)d_in[4];
    const float* outp_b    = (const float*)d_in[5];
    const float* in_proj_w = (const float*)d_in[6];
    const float* conv_w    = (const float*)d_in[7];
    const float* conv_b    = (const float*)d_in[8];
    const float* x_proj_w  = (const float*)d_in[9];
    const float* dt_proj_w = (const float*)d_in[10];
    const float* dt_proj_b = (const float*)d_in[11];
    const float* A_log     = (const float*)d_in[12];
    const float* D_param   = (const float*)d_in[13];
    const float* out_proj_w= (const float*)d_in[14];
    const float* norm_w    = (const float*)d_in[15];
    const float* norm_f_w  = (const float*)d_in[16];
    float* out = (float*)d_out;

    // workspace carve (~12 MB total)
    char* w = (char*)d_ws;
    auto carveH = [&](size_t n) { h16* p = (h16*)w; w += n * sizeof(h16); return p; };
    h16* inp16     = carveH((size_t)DM_ * IN_);
    h16* outp16    = carveH((size_t)OUT_ * DM_);
    h16* inproj16  = carveH((size_t)L_ * 2 * DI_ * DM_);
    h16* xproj16   = carveH((size_t)L_ * (DTR_ + 2 * DS_) * DI_);
    h16* dtproj16  = carveH((size_t)L_ * DI_ * DTR_);
    h16* outproj16 = carveH((size_t)L_ * DM_ * DI_);
    h16* yg        = carveH((size_t)B_ * DI_);
    auto carveF = [&](size_t n) { float* p = (float*)w; w += n * sizeof(float); return p; };
    float* Aneg   = carveF((size_t)L_ * DI_ * DS_);
    float* residA = carveF((size_t)B_ * DM_);
    float* residB = carveF((size_t)B_ * DM_);
    float* hid    = carveF((size_t)B_ * DM_);
    float* xz     = carveF((size_t)B_ * 2 * DI_);

    auto cvt = [&](const float* s, h16* d, int n) {
        cvt_f16_kernel<<<(n + 255) / 256, 256, 0, stream>>>(s, d, n);
    };
    cvt(inp_w, inp16, DM_ * IN_);
    cvt(outp_w, outp16, OUT_ * DM_);
    cvt(in_proj_w, inproj16, L_ * 2 * DI_ * DM_);
    cvt(x_proj_w, xproj16, L_ * (DTR_ + 2 * DS_) * DI_);
    cvt(dt_proj_w, dtproj16, L_ * DI_ * DTR_);
    cvt(out_proj_w, outproj16, L_ * DM_ * DI_);
    aneg_kernel<<<(L_ * DI_ * DS_ + 255) / 256, 256, 0, stream>>>(A_log, Aneg, L_ * DI_ * DS_);

    // states live in d_out (after y region), updated in place -> final states land correctly
    float* stBase = out + (size_t)T_ * B_ * OUT_;
    hipMemcpyAsync(stBase, rnn, (size_t)L_ * B_ * DI_ * (DC_ + DS_) * sizeof(float),
                   hipMemcpyDeviceToDevice, stream);

    size_t smSsm = (size_t)(16 * 1024 + 16 * 1024 + 16 * 64) * sizeof(float);  // ~132 KB

    for (int t = 0; t < T_; t++) {
        const float* xt = x + (size_t)t * B_ * IN_;
        // hidden = x_t @ inp_w^T + inp_b
        gemm_bias_kernel<float, true><<<dim3(B_ / 64, DM_ / 128), 128, 0, stream>>>(
            xt, inp16, inp_b, hid, IN_, DM_);
        for (int l = 0; l < L_; l++) {
            // residual update + rmsnorm + in_proj (residual ping-pong A -> B)
            if (l == 0)
                gemm_norm_kernel<false, true, false>
                    <<<dim3(B_ / 64, (2 * DI_) / 128), 128, 0, stream>>>(
                        hid, nullptr, residA, norm_w, inproj16, nullptr, xz, 2 * DI_);
            else
                gemm_norm_kernel<true, true, false>
                    <<<dim3(B_ / 64, (2 * DI_) / 128), 128, 0, stream>>>(
                        hid, residA, residB, norm_w + DM_, inproj16 + (size_t)2 * DI_ * DM_,
                        nullptr, xz, 2 * DI_);
            // conv + ssm + gate
            ssm_kernel<<<dim3(B_ / 16), 256, smSsm, stream>>>(
                xz, stBase + (size_t)l * B_ * DI_ * (DC_ + DS_),
                conv_w + (size_t)l * DI_ * DC_, conv_b + (size_t)l * DI_,
                xproj16 + (size_t)l * (DTR_ + 2 * DS_) * DI_,
                dtproj16 + (size_t)l * DI_ * DTR_, dt_proj_b + (size_t)l * DI_,
                Aneg + (size_t)l * DI_ * DS_, D_param + (size_t)l * DI_, yg);
            // hidden = yg @ out_proj^T  (A operand already f16)
            gemm_bias_kernel<h16, false><<<dim3(B_ / 64, DM_ / 128), 128, 0, stream>>>(
                yg, outproj16 + (size_t)l * DM_ * DI_, nullptr, hid, DI_, DM_);
        }
        // y[t] = rmsnorm(hidden + residual, norm_f_w) @ outp_w^T + outp_b
        gemm_norm_kernel<true, false, true><<<dim3(B_ / 64, OUT_ / 128), 128, 0, stream>>>(
            hid, residB, nullptr, norm_f_w, outp16, outp_b, out + (size_t)t * B_ * OUT_, OUT_);
    }
}
